// SwinTransformerBlock_71184787964118
// MI455X (gfx1250) — compile-verified, matrix operand order checked by reading, not run on hardware
//
#include <hip/hip_runtime.h>
#include <hip/hip_bf16.h>

// ---------------------------------------------------------------------------
// Swin Transformer block (shifted-window attention + MLP) for gfx1250.
// All GEMMs run on v_wmma_f32_16x16x32_f16 (f32 accumulate).
// Shapes (fixed by the reference): B=8, H=W=128, C=192, heads=6, hd=32,
// WS=8 (N=64 tokens/window), SHIFT=4, hidden=768, tokens=131072.
// ---------------------------------------------------------------------------

#define WS_     8
#define SHIFT_  4
#define HEADS_  6
#define DIM_    192
#define HD_     32
#define NTOK_   64          // tokens per window
#define TOKENS_ 131072      // B*H*W
#define NWIN_   2048        // B * (128/8)^2
#define HIDDEN_ 768

typedef _Float16 h8  __attribute__((ext_vector_type(8)));
typedef _Float16 h16 __attribute__((ext_vector_type(16)));
typedef float    f8  __attribute__((ext_vector_type(8)));

static __device__ __forceinline__ h16 make_frag(const _Float16* p0, const _Float16* p1) {
    h8 lo = *(const h8*)p0;   // 16B aligned by construction
    h8 hi = *(const h8*)p1;
    return __builtin_shufflevector(lo, hi, 0,1,2,3,4,5,6,7,8,9,10,11,12,13,14,15);
}

static __device__ __forceinline__ f8 wmma_f16(h16 a, h16 b, f8 c) {
    return __builtin_amdgcn_wmma_f32_16x16x32_f16(false, a, false, b, (short)0, c, false, false);
}

// window-order row -> original token index (window reverse + roll(+4,+4))
static __device__ __forceinline__ int win_to_orig(int row) {
    int win = row >> 6, n = row & 63;
    int b  = win >> 8, wp = win & 255;
    int wh = wp >> 4,  ww = wp & 15;
    int hs = wh * 8 + (n >> 3);
    int ws = ww * 8 + (n & 7);
    int h  = (hs + SHIFT_) & 127;
    int w  = (ws + SHIFT_) & 127;
    return (b << 14) + (h << 7) + w;
}

// original token -> window-order row (roll(-4,-4) + window partition)
static __device__ __forceinline__ int orig_to_win(int t) {
    int b  = t >> 14, rem = t & 16383;
    int hh = rem >> 7, ww = rem & 127;
    int hs = (hh + 128 - SHIFT_) & 127;
    int ws = (ww + 128 - SHIFT_) & 127;
    int win = (b << 8) + ((hs >> 3) << 4) + (ws >> 3);
    int n   = ((hs & 7) << 3) + (ws & 7);
    return win * NTOK_ + n;
}

// ---------------------------------------------------------------------------
// Weight convert + transpose: wt[n*K+k] = (f16) w[k*N+n]
// ---------------------------------------------------------------------------
__global__ void convert_wt_kernel(const float* __restrict__ w, _Float16* __restrict__ wt,
                                  int K, int N) {
    int idx = blockIdx.x * 256 + threadIdx.x;
    if (idx >= K * N) return;
    int n = idx / K, k = idx % K;
    wt[idx] = (_Float16)w[(size_t)k * N + n];
}

// ---------------------------------------------------------------------------
// LayerNorm, one wave32 per token (192 = 32 lanes * 6).
// WINDOW_DST: scatter output rows into shifted-window order.
// ---------------------------------------------------------------------------
template<bool WINDOW_DST>
__global__ __launch_bounds__(256) void ln_kernel(const float* __restrict__ x,
                                                 const float* __restrict__ w,
                                                 const float* __restrict__ b,
                                                 _Float16* __restrict__ out) {
    const int wave = threadIdx.x >> 5, lane = threadIdx.x & 31;
    const int token = blockIdx.x * 8 + wave;
    const float* xp = x + (size_t)token * DIM_;

    float v[6];
    float s = 0.f;
#pragma unroll
    for (int i = 0; i < 6; ++i) { v[i] = xp[lane + 32 * i]; s += v[i]; }
#pragma unroll
    for (int off = 16; off >= 1; off >>= 1) s += __shfl_xor(s, off, 32);
    const float mu = s * (1.0f / DIM_);

    float q = 0.f;
#pragma unroll
    for (int i = 0; i < 6; ++i) { float d = v[i] - mu; q += d * d; }
#pragma unroll
    for (int off = 16; off >= 1; off >>= 1) q += __shfl_xor(q, off, 32);
    const float rs = rsqrtf(q * (1.0f / DIM_) + 1e-5f);

    const int drow = WINDOW_DST ? orig_to_win(token) : token;
    _Float16* op = out + (size_t)drow * DIM_;
#pragma unroll
    for (int i = 0; i < 6; ++i) {
        int c = lane + 32 * i;
        op[c] = (_Float16)((v[i] - mu) * rs * w[c] + b[c]);
    }
}

// ---------------------------------------------------------------------------
// Generic WMMA GEMM: C[M,N] = A[M,K] (f16 row-major) * Wt[N,K]^T (f16) + epi.
// 256x64 C tile per 256-thread block; 8 waves, each owns a 32x64 slice:
// 2 A-fragments x 4 B-fragments -> 8 WMMA per 32-wide k-step
// (1.5 ds_load_b128 per WMMA; 2 barriers amortized over 8 WMMAs).
// Global staging for step i is issued into registers before the barrier that
// closes step i-1's compute, overlapping HBM latency with the WMMA chain.
// EPI: 0 = +bias -> f16          (QKV)
//      1 = +bias +res[x], scatter window->orig -> f32 x2   (proj)
//      2 = gelu(+bias) -> f16    (MLP fc1)
//      3 = +bias +res[x2] -> f32 (MLP fc2, final output)
// ---------------------------------------------------------------------------
template<int K, int N, int EPI>
__global__ __launch_bounds__(256) void gemm_wmma_kernel(
        const _Float16* __restrict__ A, const _Float16* __restrict__ Wt,
        const float* __restrict__ bias, const float* __restrict__ res,
        void* __restrict__ outp) {
    __shared__ alignas(16) _Float16 As[256][40];  // 80B row stride, 16B aligned
    __shared__ alignas(16) _Float16 Bs[64][40];

    const int m0 = blockIdx.y * 256;
    const int n0 = blockIdx.x * 64;
    const int tid = threadIdx.x;
    const int wave = tid >> 5, lane = tid & 31;
    const int r  = lane & 15;
    const int kb = (lane >> 4) * 8;

    const int srow = tid >> 2;          // staging row (x4 row-groups of 64)
    const int sseg = (tid & 3) * 8;     // staging k-seg {0,8,16,24}

    f8 acc[2][4];
#pragma unroll
    for (int t = 0; t < 2; ++t)
#pragma unroll
        for (int ni = 0; ni < 4; ++ni)
            acc[t][ni] = f8{0.f,0.f,0.f,0.f,0.f,0.f,0.f,0.f};

    for (int k0 = 0; k0 < K; k0 += 32) {
        // issue global loads for this step before closing previous compute
        h8 pa[4], pb;
#pragma unroll
        for (int i = 0; i < 4; ++i)
            pa[i] = *(const h8*)(A + (size_t)(m0 + srow + i * 64) * K + k0 + sseg);
        pb = *(const h8*)(Wt + (size_t)(n0 + srow) * K + k0 + sseg);

        __syncthreads();
#pragma unroll
        for (int i = 0; i < 4; ++i)
            *(h8*)&As[srow + i * 64][sseg] = pa[i];
        if (srow < 64)
            *(h8*)&Bs[srow][sseg] = pb;
        __syncthreads();

        h16 af0 = make_frag(&As[(2 * wave)     * 16 + r][kb], &As[(2 * wave)     * 16 + r][16 + kb]);
        h16 af1 = make_frag(&As[(2 * wave + 1) * 16 + r][kb], &As[(2 * wave + 1) * 16 + r][16 + kb]);
#pragma unroll
        for (int ni = 0; ni < 4; ++ni) {
            h16 bf = make_frag(&Bs[ni * 16 + r][kb], &Bs[ni * 16 + r][16 + kb]);
            acc[0][ni] = wmma_f16(af0, bf, acc[0][ni]);
            acc[1][ni] = wmma_f16(af1, bf, acc[1][ni]);
        }
    }

    // ---- epilogue ---------------------------------------------------------
#pragma unroll
    for (int t = 0; t < 2; ++t) {
        const int rowb = m0 + (2 * wave + t) * 16 + 8 * (lane >> 4);
#pragma unroll
        for (int j = 0; j < 8; ++j) {
            const int row = rowb + j;
            int orow = row;
            if constexpr (EPI == 1) orow = win_to_orig(row);
#pragma unroll
            for (int ni = 0; ni < 4; ++ni) {
                const int col = n0 + ni * 16 + (lane & 15);
                const float v = acc[t][ni][j] + bias[col];
                if constexpr (EPI == 0) {
                    ((_Float16*)outp)[(size_t)row * N + col] = (_Float16)v;
                } else if constexpr (EPI == 1) {
                    float* o = (float*)outp;
                    const size_t d = (size_t)orow * N + col;
                    o[d] = res[d] + v;
                } else if constexpr (EPI == 2) {
                    const float g = 0.5f * v * (1.0f + erff(v * 0.70710678118654752f));
                    ((_Float16*)outp)[(size_t)row * N + col] = (_Float16)g;
                } else {
                    float* o = (float*)outp;
                    const size_t d = (size_t)row * N + col;
                    o[d] = res[d] + v;
                }
            }
        }
    }
}

// ---------------------------------------------------------------------------
// Windowed attention: one block per (head, window). 128 threads = 4 waves,
// each wave owns a 16-row slice of the 64x64 score matrix.
// qkv layout: [win*64+n][576] f16, q at +0, k at +192, v at +384 (per head *32).
// ---------------------------------------------------------------------------
__global__ __launch_bounds__(128) void attn_kernel(const _Float16* __restrict__ qkv,
                                                   const float* __restrict__ rpb,
                                                   _Float16* __restrict__ out) {
    __shared__ float               Sf[64][65];
    __shared__ alignas(16) _Float16 P[64][72];
    __shared__ alignas(16) _Float16 Vt[32][72];

    const int head = blockIdx.x;
    const int win  = blockIdx.y;
    const int tid  = threadIdx.x;
    const int wave = tid >> 5, lane = tid & 31;
    const int r  = lane & 15;
    const int kb = (lane >> 4) * 8;
    const size_t rowbase = (size_t)win * NTOK_;
    const int QC = 3 * DIM_;

    // ---- S = q @ k^T ------------------------------------------------------
    const _Float16* qp = qkv + (rowbase + wave * 16 + r) * QC + head * HD_;
    h16 qf = make_frag(qp + kb, qp + 16 + kb);

    f8 s[4];
#pragma unroll
    for (int mt = 0; mt < 4; ++mt) {
        const _Float16* kp = qkv + (rowbase + mt * 16 + r) * QC + DIM_ + head * HD_;
        h16 kf = make_frag(kp + kb, kp + 16 + kb);
        f8 z = {0.f,0.f,0.f,0.f,0.f,0.f,0.f,0.f};
        s[mt] = wmma_f16(qf, kf, z);
    }

    // ---- scale + rel-pos bias + shift mask -> LDS (fp32) ------------------
    const float scale = 0.17677669529663687f;   // 1/sqrt(32)
    const int wp = win & 255;
    const int wh = wp >> 4, ww = wp & 15;
    const bool edge = (wh == 15) || (ww == 15);
#pragma unroll
    for (int mt = 0; mt < 4; ++mt) {
#pragma unroll
        for (int j = 0; j < 8; ++j) {
            const int n_ = wave * 16 + j + 8 * (lane >> 4);
            const int m_ = mt * 16 + (lane & 15);
            const int ih = n_ >> 3, iw = n_ & 7;
            const int jh = m_ >> 3, jw = m_ & 7;
            const int idx = (ih - jh + 7) * 15 + (iw - jw + 7);
            float val = s[mt][j] * scale + rpb[idx * HEADS_ + head];
            if (edge) {
                const int ch0 = wh * 8 + ih, ch1 = wh * 8 + jh;
                const int cw0 = ww * 8 + iw, cw1 = ww * 8 + jw;
                const int r0 = (ch0 < 120 ? 0 : (ch0 < 124 ? 1 : 2)) * 3 +
                               (cw0 < 120 ? 0 : (cw0 < 124 ? 1 : 2));
                const int r1 = (ch1 < 120 ? 0 : (ch1 < 124 ? 1 : 2)) * 3 +
                               (cw1 < 120 ? 0 : (cw1 < 124 ? 1 : 2));
                if (r0 != r1) val += -100.0f;
            }
            Sf[n_][m_] = val;
        }
    }
    __syncthreads();

    // ---- stage V^T (all threads) + row softmax (threads 0..63) ------------
    {
        const int t = tid >> 1;
        const int d0 = (tid & 1) * 16;
        const _Float16* vp = qkv + (rowbase + t) * QC + 2 * DIM_ + head * HD_ + d0;
#pragma unroll
        for (int dd = 0; dd < 16; ++dd) Vt[d0 + dd][t] = vp[dd];
    }
    if (tid < 64) {
        float mx = -1e30f;
#pragma unroll 8
        for (int m = 0; m < 64; ++m) mx = fmaxf(mx, Sf[tid][m]);
        float sum = 0.f;
#pragma unroll 8
        for (int m = 0; m < 64; ++m) { float e = __expf(Sf[tid][m] - mx); Sf[tid][m] = e; sum += e; }
        const float inv = 1.0f / sum;
#pragma unroll 8
        for (int m = 0; m < 64; ++m) P[tid][m] = (_Float16)(Sf[tid][m] * inv);
    }
    __syncthreads();

    // ---- O = P @ V --------------------------------------------------------
    f8 o0 = {0.f,0.f,0.f,0.f,0.f,0.f,0.f,0.f};
    f8 o1 = {0.f,0.f,0.f,0.f,0.f,0.f,0.f,0.f};
#pragma unroll
    for (int kk = 0; kk < 64; kk += 32) {
        h16 pf = make_frag(&P[wave * 16 + r][kk + kb], &P[wave * 16 + r][kk + 16 + kb]);
        h16 v0 = make_frag(&Vt[(lane & 15)][kk + kb],      &Vt[(lane & 15)][kk + 16 + kb]);
        h16 v1 = make_frag(&Vt[16 + (lane & 15)][kk + kb], &Vt[16 + (lane & 15)][kk + 16 + kb]);
        o0 = wmma_f16(pf, v0, o0);
        o1 = wmma_f16(pf, v1, o1);
    }
#pragma unroll
    for (int j = 0; j < 8; ++j) {
        const int n_ = wave * 16 + j + 8 * (lane >> 4);
        _Float16* op = out + (rowbase + n_) * DIM_ + head * HD_;
        op[(lane & 15)]      = (_Float16)o0[j];
        op[16 + (lane & 15)] = (_Float16)o1[j];
    }
}

// ---------------------------------------------------------------------------
// Host-side orchestration
// ---------------------------------------------------------------------------
extern "C" void kernel_launch(void* const* d_in, const int* in_sizes, int n_in,
                              void* d_out, int out_size, void* d_ws, size_t ws_size,
                              hipStream_t stream) {
    (void)in_sizes; (void)n_in; (void)out_size; (void)ws_size;

    const float* x       = (const float*)d_in[0];
    const float* qkv_w   = (const float*)d_in[3];
    const float* qkv_b   = (const float*)d_in[4];
    const float* proj_w  = (const float*)d_in[5];
    const float* proj_b  = (const float*)d_in[6];
    const float* rpb     = (const float*)d_in[7];
    const float* n1w     = (const float*)d_in[8];
    const float* n1b     = (const float*)d_in[9];
    const float* n2w     = (const float*)d_in[10];
    const float* n2b     = (const float*)d_in[11];
    const float* mlp_w1  = (const float*)d_in[12];
    const float* mlp_b1  = (const float*)d_in[13];
    const float* mlp_w2  = (const float*)d_in[14];
    const float* mlp_b2  = (const float*)d_in[15];
    float* out = (float*)d_out;

    // workspace layout (bytes); regions reused across phases
    char* ws = (char*)d_ws;
    const size_t szBig  = (size_t)TOKENS_ * HIDDEN_ * 2;   // 201,326,592 (qkv then mlp hidden)
    const size_t szAct  = (size_t)TOKENS_ * DIM_ * 2;      //  50,331,648
    const size_t szX2   = (size_t)TOKENS_ * DIM_ * 4;      // 100,663,296
    _Float16* buf_big  = (_Float16*)(ws);                          // qkv / hidden
    _Float16* buf_ln   = (_Float16*)(ws + szBig);                  // ln1(window) / h2
    _Float16* buf_attn = (_Float16*)(ws + szBig + szAct);          // attention out
    float*    buf_x2   = (float*)   (ws + szBig + 2 * szAct);      // x2 residual
    char*     wbase    = ws + szBig + 2 * szAct + szX2;
    _Float16* wt_qkv  = (_Float16*)(wbase);
    _Float16* wt_proj = (_Float16*)(wbase + 221184);
    _Float16* wt_m1   = (_Float16*)(wbase + 221184 + 73728);
    _Float16* wt_m2   = (_Float16*)(wbase + 221184 + 73728 + 294912);

    // 1) convert + transpose weights to f16 [N][K]
    convert_wt_kernel<<<(DIM_ * 3 * DIM_ + 255) / 256, 256, 0, stream>>>(qkv_w, wt_qkv, DIM_, 3 * DIM_);
    convert_wt_kernel<<<(DIM_ * DIM_ + 255) / 256, 256, 0, stream>>>(proj_w, wt_proj, DIM_, DIM_);
    convert_wt_kernel<<<(DIM_ * HIDDEN_ + 255) / 256, 256, 0, stream>>>(mlp_w1, wt_m1, DIM_, HIDDEN_);
    convert_wt_kernel<<<(HIDDEN_ * DIM_ + 255) / 256, 256, 0, stream>>>(mlp_w2, wt_m2, HIDDEN_, DIM_);

    // 2) LN1 -> f16 in shifted-window token order
    ln_kernel<true><<<TOKENS_ / 8, 256, 0, stream>>>(x, n1w, n1b, buf_ln);

    // 3) QKV GEMM: [131072,192] x [192,576] + bias -> f16
    gemm_wmma_kernel<DIM_, 3 * DIM_, 0><<<dim3(9, TOKENS_ / 256), 256, 0, stream>>>(
        buf_ln, wt_qkv, qkv_b, nullptr, buf_big);

    // 4) windowed attention (rel-pos bias + shift mask + softmax + P@V)
    attn_kernel<<<dim3(HEADS_, NWIN_), 128, 0, stream>>>(buf_big, rpb, buf_attn);

    // 5) proj GEMM + window-reverse/roll scatter + residual -> x2 (f32)
    gemm_wmma_kernel<DIM_, DIM_, 1><<<dim3(3, TOKENS_ / 256), 256, 0, stream>>>(
        buf_attn, wt_proj, proj_b, x, buf_x2);

    // 6) LN2 -> f16 (straight token order)
    ln_kernel<false><<<TOKENS_ / 8, 256, 0, stream>>>(buf_x2, n2w, n2b, buf_ln);

    // 7) MLP fc1 + exact GELU -> f16
    gemm_wmma_kernel<DIM_, HIDDEN_, 2><<<dim3(12, TOKENS_ / 256), 256, 0, stream>>>(
        buf_ln, wt_m1, mlp_b1, nullptr, buf_big);

    // 8) MLP fc2 + bias + residual(x2) -> final f32 output
    gemm_wmma_kernel<HIDDEN_, DIM_, 3><<<dim3(3, TOKENS_ / 256), 256, 0, stream>>>(
        buf_big, wt_m2, mlp_b2, buf_x2, out);
}
